// HGCN_32212254720551
// MI455X (gfx1250) — matrix operand order repaired and checked by reading, not compiled, not run on hardware
//
#include <hip/hip_runtime.h>

#define N_NODES  100000
#define N_EDGES  20000
#define NNZ      600000
#define N_GRAPHS 64
#define CCH      128
#define OUTC     16

typedef __attribute__((ext_vector_type(8)))  _Float16 v8h;
typedef __attribute__((ext_vector_type(16))) _Float16 v16h;
typedef __attribute__((ext_vector_type(8)))  float    v8f;

union AFrag { v16h v; v8h h[2]; };

__device__ __forceinline__ void atomAddF(float* p, float v) {
  unsafeAtomicAdd(p, v);   // lowers to global_atomic_add_f32 (no CAS loop)
}

// ---------------------------------------------------------------- utilities
__global__ __launch_bounds__(256) void k_zero4(float4* __restrict__ p, int n4) {
  int i = blockIdx.x * 256 + threadIdx.x;
  if (i < n4) p[i] = make_float4(0.f, 0.f, 0.f, 0.f);
}

__global__ __launch_bounds__(256) void k_count_deg(const int* __restrict__ nidx,
                                                   const int* __restrict__ eidx,
                                                   float* __restrict__ Dn,
                                                   float* __restrict__ De) {
  int i = blockIdx.x * 256 + threadIdx.x;
  if (i < NNZ) {
    atomAddF(&Dn[nidx[i]], 1.0f);
    atomAddF(&De[eidx[i]], 1.0f);
  }
}

__global__ __launch_bounds__(256) void k_invert(float* __restrict__ p, int n) {
  int i = blockIdx.x * 256 + threadIdx.x;
  if (i < n) { float v = p[i]; p[i] = v > 0.f ? 1.f / v : 0.f; }
}

// Pack 128x128 f32 row-major W into per-lane WMMA B-fragment layout (f16).
// Layout index: ((kstep*8 + ntile)*32 + lane)*16 + j
//   lane < 16: holds column n = ntile*16 + lane,      K = kstep*32 + (j<8 ? j : 16+j-8)
//   lane >=16: holds column n = ntile*16 + lane - 16, K = kstep*32 + 8 + (j<8 ? j : 16+j-8)
__global__ __launch_bounds__(256) void k_pack_w(const float* __restrict__ W,
                                                _Float16* __restrict__ Wp) {
  int o    = blockIdx.x * 256 + threadIdx.x;   // 16384 elements
  int j    = o & 15;
  int lane = (o >> 4) & 31;
  int n0   = (o >> 9) & 7;
  int ks   = o >> 12;
  int k = ks * 32 + ((lane < 16) ? 0 : 8) + (j & 7) + ((j & 8) << 1);
  int n = n0 * 16 + (lane & 15);
  Wp[o] = (_Float16)W[k * CCH + n];
}

// Fused: edge-feature scale by B^{-1} + f32 -> f16 convert (GEMM A staging).
__global__ __launch_bounds__(256) void k_pack_a(const float* __restrict__ src,
                                                const float* __restrict__ rscale,
                                                _Float16* __restrict__ dst, int n) {
  int i = blockIdx.x * 256 + threadIdx.x;
  if (i < n) dst[i] = (_Float16)(src[i] * rscale[i >> 7]);
}

// One wave moves one 128-float row: gather float4/lane, atomic-add to dst row.
__global__ __launch_bounds__(256) void k_scatter_rows(const float* __restrict__ src,
                                                      const int* __restrict__ gidx,
                                                      const int* __restrict__ sidx,
                                                      float* __restrict__ dst, int nnz) {
  int lane = threadIdx.x & 31;
  int i = blockIdx.x * 8 + (threadIdx.x >> 5);
  if (i >= nnz) return;
  int g = gidx[i], s = sidx[i];
  float4 v = *(const float4*)(src + g * CCH + lane * 4);
  float* d = dst + s * CCH + lane * 4;
  atomAddF(d + 0, v.x); atomAddF(d + 1, v.y);
  atomAddF(d + 2, v.z); atomAddF(d + 3, v.w);
}

// WMMA GEMM: C[M x 128] = A[M x 128] * W[128 x 128]; M multiple of 16.
// 256 threads = 8 waves; wave w owns the 16x16 tile at columns [16w, 16w+16).
__global__ __launch_bounds__(256) void k_gemm_wmma(const _Float16* __restrict__ A,
                                                   const _Float16* __restrict__ Bp,
                                                   float* __restrict__ C) {
  int lane = threadIdx.x & 31;
  int wv   = threadIdx.x >> 5;
  int row_base = blockIdx.x << 4;
  int r  = lane & 15;                 // A: both lane halves hold rows 0..15
  int kb = (lane < 16) ? 0 : 8;       // A: K-half select per ISA 16-bit A layout
  v8f c = {};
  const _Float16* arow = A + (row_base + r) * CCH + kb;
  const _Float16* bp   = Bp + (wv << 9) + (lane << 4);
#pragma unroll
  for (int ks = 0; ks < 4; ++ks) {    // K = 128 in 4 steps of 32
    AFrag a;
    a.h[0] = *(const v8h*)(arow + ks * 32);        // K = ks*32 + kb + [0..7]
    a.h[1] = *(const v8h*)(arow + ks * 32 + 16);   // K = ks*32 + kb + [16..23]
    v16h b = *(const v16h*)(bp + (ks << 12));      // packed, 32B contiguous
    c = __builtin_amdgcn_wmma_f32_16x16x32_f16(false, a.v, false, b,
                                               (short)0, c, false, false);
  }
  // C layout: VGPR rr -> row rr (lanes 0-15) / row 8+rr (lanes 16-31), col = lane&15
  float* cp = C + (row_base + ((lane >> 4) << 3)) * CCH + (wv << 4) + r;
#pragma unroll
  for (int rr = 0; rr < 8; ++rr) cp[rr * CCH] = c[rr];
}

// h = relu(h * Dinv[row] + b[col])
__global__ __launch_bounds__(256) void k_finalize(float* __restrict__ h,
                                                  const float* __restrict__ Dinv,
                                                  const float* __restrict__ b, int n) {
  int i = blockIdx.x * 256 + threadIdx.x;
  if (i < n) {
    float v = h[i] * Dinv[i >> 7] + b[i & 127];
    h[i] = v > 0.f ? v : 0.f;
  }
}

__global__ __launch_bounds__(256) void k_pool(const float* __restrict__ h,
                                              const int* __restrict__ batch,
                                              float* __restrict__ pooled,
                                              float* __restrict__ cnt) {
  int lane = threadIdx.x & 31;
  int i = blockIdx.x * 8 + (threadIdx.x >> 5);
  if (i >= N_NODES) return;
  int g = batch[i];
  float4 v = *(const float4*)(h + i * CCH + lane * 4);
  float* d = pooled + g * CCH + lane * 4;
  atomAddF(d + 0, v.x); atomAddF(d + 1, v.y);
  atomAddF(d + 2, v.z); atomAddF(d + 3, v.w);
  if (lane == 0) atomAddF(&cnt[g], 1.0f);
}

// 64 threads: per-graph mean, 128x16 matvec, log_softmax.
__global__ __launch_bounds__(64) void k_head(const float* __restrict__ pooled,
                                             const float* __restrict__ cnt,
                                             const float* __restrict__ Wfc,
                                             const float* __restrict__ bfc,
                                             float* __restrict__ out) {
  int g = threadIdx.x;
  if (g >= N_GRAPHS) return;
  float inv = 1.0f / fmaxf(cnt[g], 1.0f);
  float lg[OUTC];
#pragma unroll
  for (int c = 0; c < OUTC; ++c) lg[c] = bfc[c];
  for (int k = 0; k < CCH; ++k) {
    float pv = pooled[g * CCH + k] * inv;
#pragma unroll
    for (int c = 0; c < OUTC; ++c) lg[c] += pv * Wfc[k * OUTC + c];
  }
  float mx = lg[0];
#pragma unroll
  for (int c = 1; c < OUTC; ++c) mx = fmaxf(mx, lg[c]);
  float s = 0.f;
#pragma unroll
  for (int c = 0; c < OUTC; ++c) s += __expf(lg[c] - mx);
  float lse = __logf(s);
#pragma unroll
  for (int c = 0; c < OUTC; ++c) out[g * OUTC + c] = lg[c] - mx - lse;
}

// ---------------------------------------------------------------- host side
extern "C" void kernel_launch(void* const* d_in, const int* in_sizes, int n_in,
                              void* d_out, int out_size, void* d_ws, size_t ws_size,
                              hipStream_t stream) {
  (void)in_sizes; (void)n_in; (void)out_size; (void)ws_size;
  const float* x        = (const float*)d_in[0];
  const int*   node_idx = (const int*)d_in[1];
  const int*   edge_idx = (const int*)d_in[2];
  const int*   batch    = (const int*)d_in[3];
  const float* W1       = (const float*)d_in[4];
  const float* b1       = (const float*)d_in[5];
  const float* W2       = (const float*)d_in[6];
  const float* b2       = (const float*)d_in[7];
  const float* Wfc      = (const float*)d_in[8];
  const float* bfc      = (const float*)d_in[9];
  float* out = (float*)d_out;

  // bump allocator over workspace (256B aligned blocks)
  char* ws = (char*)d_ws;
  size_t off = 0;
  auto alloc = [&](size_t bytes) -> void* {
    void* p = ws + off;
    off = (off + bytes + 255) & ~(size_t)255;
    return p;
  };
  float*    DnInv  = (float*)alloc((size_t)N_NODES * 4);
  float*    BeInv  = (float*)alloc((size_t)N_EDGES * 4);
  float*    aggE   = (float*)alloc((size_t)N_EDGES * CCH * 4);   // 10.24 MB
  float*    eBuf   = (float*)alloc((size_t)N_EDGES * CCH * 4);   // 10.24 MB
  float*    hBuf   = (float*)alloc((size_t)N_NODES * CCH * 4);   // 51.2 MB
  _Float16* A16    = (_Float16*)alloc((size_t)N_EDGES * CCH * 2);
  _Float16* W1p    = (_Float16*)alloc((size_t)CCH * CCH * 2);
  _Float16* W2p    = (_Float16*)alloc((size_t)CCH * CCH * 2);
  float*    pooled = (float*)alloc((size_t)N_GRAPHS * CCH * 4);
  float*    cnt    = (float*)alloc((size_t)N_GRAPHS * 4);

  const int nE  = N_EDGES * CCH;      // 2,560,000
  const int nN  = N_NODES * CCH;      // 12,800,000
  const int gbE = nE / 4 / 256;       // zero grid for edge buffers
  const int gbN = nN / 4 / 256;       // zero grid for node buffer
  const int gScat = (NNZ + 7) / 8;    // 8 rows per 256-thread block
  const int gGemE = N_EDGES / 16;     // 1250 row-tiles

  // degrees -> inverses
  k_zero4<<<N_NODES / 4 / 256 + 1, 256, 0, stream>>>((float4*)DnInv, N_NODES / 4);
  k_zero4<<<N_EDGES / 4 / 256 + 1, 256, 0, stream>>>((float4*)BeInv, N_EDGES / 4);
  k_count_deg<<<(NNZ + 255) / 256, 256, 0, stream>>>(node_idx, edge_idx, DnInv, BeInv);
  k_invert<<<(N_NODES + 255) / 256, 256, 0, stream>>>(DnInv, N_NODES);
  k_invert<<<(N_EDGES + 255) / 256, 256, 0, stream>>>(BeInv, N_EDGES);

  // pack weights once into WMMA B layout
  k_pack_w<<<64, 256, 0, stream>>>(W1, W1p);
  k_pack_w<<<64, 256, 0, stream>>>(W2, W2p);

  // -------- layer 1: agg(x) -> scale -> GEMM(W1) -> scatter -> Dinv+b1+relu
  k_zero4<<<gbE, 256, 0, stream>>>((float4*)aggE, nE / 4);
  k_scatter_rows<<<gScat, 256, 0, stream>>>(x, node_idx, edge_idx, aggE, NNZ);
  k_pack_a<<<nE / 256, 256, 0, stream>>>(aggE, BeInv, A16, nE);
  k_gemm_wmma<<<gGemE, 256, 0, stream>>>(A16, W1p, eBuf);
  k_zero4<<<gbN, 256, 0, stream>>>((float4*)hBuf, nN / 4);
  k_scatter_rows<<<gScat, 256, 0, stream>>>(eBuf, edge_idx, node_idx, hBuf, NNZ);
  k_finalize<<<nN / 256, 256, 0, stream>>>(hBuf, DnInv, b1, nN);

  // -------- layer 2: same pipeline on h
  k_zero4<<<gbE, 256, 0, stream>>>((float4*)aggE, nE / 4);
  k_scatter_rows<<<gScat, 256, 0, stream>>>(hBuf, node_idx, edge_idx, aggE, NNZ);
  k_pack_a<<<nE / 256, 256, 0, stream>>>(aggE, BeInv, A16, nE);
  k_gemm_wmma<<<gGemE, 256, 0, stream>>>(A16, W2p, eBuf);
  k_zero4<<<gbN, 256, 0, stream>>>((float4*)hBuf, nN / 4);
  k_scatter_rows<<<gScat, 256, 0, stream>>>(eBuf, edge_idx, node_idx, hBuf, NNZ);
  k_finalize<<<nN / 256, 256, 0, stream>>>(hBuf, DnInv, b2, nN);

  // -------- pool + head
  k_zero4<<<(N_GRAPHS * CCH / 4 + 255) / 256, 256, 0, stream>>>((float4*)pooled,
                                                                N_GRAPHS * CCH / 4);
  k_zero4<<<1, 256, 0, stream>>>((float4*)cnt, N_GRAPHS / 4);
  k_pool<<<(N_NODES + 7) / 8, 256, 0, stream>>>(hBuf, batch, pooled, cnt);
  k_head<<<1, 64, 0, stream>>>(pooled, cnt, Wfc, bfc, out);
}
